// PGN_29025388987178
// MI455X (gfx1250) — matrix operand order repaired and checked by reading, not compiled, not run on hardware
//
#include <hip/hip_runtime.h>
#include <hip/hip_bf16.h>

// ---- problem constants (match reference setup_inputs) ----
#define B_    2
#define N_    512
#define F_    128
#define MID_  128
#define OUT_  128
#define JT    16              // j-tile width per workgroup
#define CHUNK 32              // senders (i) per workgroup
#define NCHUNK (N_ / CHUNK)   // 16 i-chunks
#define BIGNUM 1000000.0f

typedef __attribute__((ext_vector_type(16))) __bf16       v16bf;
typedef __attribute__((ext_vector_type(8)))  float        v8f;
typedef __attribute__((ext_vector_type(4)))  float        f32x4;
typedef __attribute__((ext_vector_type(4)))  unsigned int u32x4;

union AView { v16bf v; u32x4 u[2]; };

// ---------------------------------------------------------------------------
// Kernel A: per-(b,n) precompute (tiny GEMMs, ~0.2 GFLOP total).
//   pre1 = z@W1 + b1 + graph@Wg + bg   (constant over the max -> hoisted)
//   pre2 = z@W2 + b2                   (sender term, lives inside the max)
//   h1   = z@Wo1 + bo1
// ---------------------------------------------------------------------------
__global__ void __launch_bounds__(256) pgn_pre(
    const float* __restrict__ node, const float* __restrict__ hidden,
    const float* __restrict__ graph,
    const float* __restrict__ W1,  const float* __restrict__ b1,
    const float* __restrict__ W2,  const float* __restrict__ b2,
    const float* __restrict__ Wg,  const float* __restrict__ bg,
    const float* __restrict__ Wo1, const float* __restrict__ bo1,
    float* __restrict__ pre1, float* __restrict__ pre2, float* __restrict__ h1)
{
    const int idx = blockIdx.x * blockDim.x + threadIdx.x;   // (b*N+n)*MID + m
    const int m   = idx & (MID_ - 1);
    const int row = idx >> 7;                                 // b*N + n
    const int b   = row >> 9;
    const float* nrow = node   + (size_t)row * F_;
    const float* hrow = hidden + (size_t)row * F_;
    const float* grow = graph  + (size_t)b   * F_;
    float s1 = b1[m], s2 = b2[m], so = bo1[m], sg = bg[m];
    for (int k = 0; k < F_; ++k) {
        const float zn = nrow[k], zh = hrow[k];
        s1 += zn * W1 [k * MID_ + m] + zh * W1 [(k + F_) * MID_ + m];
        s2 += zn * W2 [k * MID_ + m] + zh * W2 [(k + F_) * MID_ + m];
        so += zn * Wo1[k * OUT_ + m] + zh * Wo1[(k + F_) * OUT_ + m];
        sg += grow[k] * Wg[k * MID_ + m];
    }
    pre1[idx] = s1 + sg;
    pre2[idx] = s2;
    h1[idx]   = so;
}

// ---------------------------------------------------------------------------
// Kernel B (dominant): streamed edge GEMM with fused masked max-reduction.
//   grid = (N/JT, NCHUNK, B) = (32,16,2) -> 1024 WGs of 8 waves (wave32).
//   Wave w owns m-columns [16w,16w+16); We fragments are converted f32->bf16
//   once and pinned in VGPRs in the ISA 32x16 bf16 B-layout.
//   Per sender i: 16(j) x 128(K) f32 edge tile -> LDS bf16 (272B padded row
//   stride, conflict-free b128 reads), 4 chained v_wmma_f32_16x16x32_bf16
//   (K=128), epilogue: + pre2[b,i,m] + be[m], adjacency mask, register fmax.
// ---------------------------------------------------------------------------
__global__ void __launch_bounds__(256) pgn_edge_wmma(
    const float* __restrict__ edge, const float* __restrict__ adj,
    const float* __restrict__ We,   const float* __restrict__ be,
    const float* __restrict__ pre2, float* __restrict__ partial)
{
    __shared__ __align__(16) __bf16 Atile[16 * 136];  // 16 rows, 136 bf16 stride
    __shared__ float adjT[16];

    const int t      = threadIdx.x;
    const int lane   = t & 31;
    const int wave   = t >> 5;
    const int lanelo = lane & 15;
    const int laneHi = lane >> 4;            // 0: lanes 0-15, 1: lanes 16-31
    const int ncol   = wave * 16 + lanelo;   // output column m

    const int J0 = blockIdx.x * JT;
    const int i0 = blockIdx.y * CHUNK;
    const int b  = blockIdx.z;

    // --- B fragments: We[k][m], bf16 32x16 layout, held in VGPRs ---
    v16bf Bf[4];
    for (int k32 = 0; k32 < 4; ++k32) {
        v16bf bb;
        const int kb = k32 * 32 + laneHi * 8;
        #pragma unroll
        for (int e = 0; e < 16; ++e)
            bb[e] = (__bf16)We[(kb + e + (e >= 8 ? 8 : 0)) * MID_ + ncol];
        Bf[k32] = bb;
    }
    const float bev = be[ncol];

    float acc[8];
    #pragma unroll
    for (int r = 0; r < 8; ++r) acc[r] = -BIGNUM;    // matches -BIG_NUMBER init

    // staging role of this thread: row srow (j within tile), 8-float k-chunk
    const int srow = t >> 4;
    const int sk   = (t & 15) * 8;

    // software-pipelined prefetch of the first tile
    size_t tbase = ((size_t)(b * N_ + i0) * N_ + J0) * F_;
    f32x4 f0 = *(const f32x4*)(edge + tbase + (size_t)srow * F_ + sk);
    f32x4 f1 = *(const f32x4*)(edge + tbase + (size_t)srow * F_ + sk + 4);
    float adv = (t < 16) ? adj[(size_t)(b * N_ + i0) * N_ + J0 + t] : 0.f;

    for (int ii = 0; ii < CHUNK; ++ii) {
        const int i = i0 + ii;
        __syncthreads();                     // prior iteration's LDS reads done
        #pragma unroll
        for (int q = 0; q < 4; ++q) Atile[srow * 136 + sk + q]     = (__bf16)f0[q];
        #pragma unroll
        for (int q = 0; q < 4; ++q) Atile[srow * 136 + sk + 4 + q] = (__bf16)f1[q];
        if (t < 16) adjT[t] = adv;
        __syncthreads();                     // tile visible to all waves

        // issue next tile's global loads so VMEM overlaps the WMMA chain
        if (ii + 1 < CHUNK) {
            const size_t tn = ((size_t)(b * N_ + i + 1) * N_ + J0) * F_;
            f0 = *(const f32x4*)(edge + tn + (size_t)srow * F_ + sk);
            f1 = *(const f32x4*)(edge + tn + (size_t)srow * F_ + sk + 4);
            if (t < 16) adv = adj[(size_t)(b * N_ + i + 1) * N_ + J0 + t];
        }

        // D = A(16x128) x We-tile(128x16), K chained through the accumulator
        v8f d = {0.f, 0.f, 0.f, 0.f, 0.f, 0.f, 0.f, 0.f};
        #pragma unroll
        for (int k32 = 0; k32 < 4; ++k32) {
            AView av;
            const int base = lanelo * 136 + k32 * 32 + laneHi * 8; // 16B aligned
            av.u[0] = *(const u32x4*)(&Atile[base]);       // K = k32*32 + hi*8 + [0..7]
            av.u[1] = *(const u32x4*)(&Atile[base + 16]);  // K = ... + 16 + [0..7]
            d = __builtin_amdgcn_wmma_f32_16x16x32_bf16(
                    false, av.v, false, Bf[k32], (short)0, d, false, false);
        }

        // fused epilogue: + msg_2[b,i,m] + be[m], adjacency mask, running max
        const float p2 = pre2[(size_t)(b * N_ + i) * MID_ + ncol] + bev;
        #pragma unroll
        for (int r = 0; r < 8; ++r) {
            const int jrow = laneHi * 8 + r;           // D row: M = r or 8+r
            const float v = d[r] + p2;
            if (adjT[jrow] > 0.f) acc[r] = fmaxf(acc[r], v);
        }
    }

    // partial[chunk][b][j][m]
    #pragma unroll
    for (int r = 0; r < 8; ++r) {
        const int jrow = laneHi * 8 + r;
        partial[(((size_t)blockIdx.y * B_ + b) * N_ + (J0 + jrow)) * MID_ + ncol] = acc[r];
    }
}

// ---------------------------------------------------------------------------
// Kernel C1: max across i-chunks, add hoisted pre1 (= msg_1 + msg_g).
// ---------------------------------------------------------------------------
__global__ void __launch_bounds__(256) pgn_reduce(
    const float* __restrict__ partial, const float* __restrict__ pre1,
    float* __restrict__ msred)
{
    const int idx = blockIdx.x * blockDim.x + threadIdx.x;
    float r = partial[idx];
    for (int c = 1; c < NCHUNK; ++c)
        r = fmaxf(r, partial[(size_t)c * (B_ * N_ * MID_) + idx]);
    msred[idx] = r + pre1[idx];
}

// ---------------------------------------------------------------------------
// Kernel C2: out = relu(h1 + msred @ Wo2 + bo2)   (tiny, 34 MFLOP)
// ---------------------------------------------------------------------------
__global__ void __launch_bounds__(256) pgn_out(
    const float* __restrict__ msred, const float* __restrict__ h1,
    const float* __restrict__ Wo2,   const float* __restrict__ bo2,
    float* __restrict__ out)
{
    const int idx = blockIdx.x * blockDim.x + threadIdx.x;
    const int m   = idx & (OUT_ - 1);
    const int row = idx >> 7;
    float s = h1[idx] + bo2[m];
    const float* mr = msred + (size_t)row * MID_;
    for (int k = 0; k < MID_; ++k) s += mr[k] * Wo2[k * OUT_ + m];
    out[idx] = fmaxf(s, 0.f);
}

// ---------------------------------------------------------------------------
extern "C" void kernel_launch(void* const* d_in, const int* in_sizes, int n_in,
                              void* d_out, int out_size, void* d_ws, size_t ws_size,
                              hipStream_t stream)
{
    (void)in_sizes; (void)n_in; (void)out_size; (void)ws_size;

    const float* node   = (const float*)d_in[0];
    const float* edge   = (const float*)d_in[1];
    const float* graph  = (const float*)d_in[2];
    const float* adj    = (const float*)d_in[3];
    const float* hidden = (const float*)d_in[4];
    const float* W1  = (const float*)d_in[5];  const float* b1  = (const float*)d_in[6];
    const float* W2  = (const float*)d_in[7];  const float* b2  = (const float*)d_in[8];
    const float* We  = (const float*)d_in[9];  const float* be  = (const float*)d_in[10];
    const float* Wg  = (const float*)d_in[11]; const float* bg  = (const float*)d_in[12];
    const float* Wo1 = (const float*)d_in[13]; const float* bo1 = (const float*)d_in[14];
    const float* Wo2 = (const float*)d_in[15]; const float* bo2 = (const float*)d_in[16];
    float* out = (float*)d_out;

    // workspace carve-up (floats): 4*131072 + 16*131072 = 10 MB
    const size_t RM = (size_t)B_ * N_ * MID_;   // 131072
    float* ws      = (float*)d_ws;
    float* pre1    = ws;
    float* pre2    = ws + RM;
    float* h1      = ws + 2 * RM;
    float* msred   = ws + 3 * RM;
    float* partial = ws + 4 * RM;               // NCHUNK * RM floats

    pgn_pre<<<dim3((unsigned)(RM / 256)), 256, 0, stream>>>(
        node, hidden, graph, W1, b1, W2, b2, Wg, bg, Wo1, bo1, pre1, pre2, h1);

    pgn_edge_wmma<<<dim3(N_ / JT, NCHUNK, B_), 256, 0, stream>>>(
        edge, adj, We, be, pre2, partial);

    pgn_reduce<<<dim3((unsigned)(RM / 256)), 256, 0, stream>>>(partial, pre1, msred);

    pgn_out<<<dim3((unsigned)(RM / 256)), 256, 0, stream>>>(msred, h1, Wo2, bo2, out);
}
